// GNN1D_L_Power_86878598464002
// MI455X (gfx1250) — matrix operand order repaired and checked by reading, not compiled, not run on hardware
//
#include <hip/hip_runtime.h>
#include <hip/hip_bf16.h>
#include <cstdint>

// ---------------- problem constants ----------------
#define Bsz   64
#define Rsz   32
#define Lsz   64
#define Ksz   16
#define Usz   4
#define Nsmp  (Bsz * Rsz)          // 2048
#define NCOL  ((size_t)Nsmp * Lsz) // 131072 columns (n*64 + l)
#define NCTM  8192                 // NCOL/16: column tiles of main activations
#define C0    128                  // 2*K*U
#define C1    256
#define C2    256
#define C3    16
#define BN_EPS 1e-5f

typedef __bf16  bf16_t;
typedef __bf16  v16bf __attribute__((ext_vector_type(16)));
typedef float   v8f   __attribute__((ext_vector_type(8)));

union FragF8 { v8f v; float e[8]; };

// Tiled (WMMA-B-fragment-major) activation layout:
// matrix [C x ncols] stored as k-tiles of 32 rows x 16 cols, 1KB each,
// inner order = lane-major: lane' = 16*(row>=16) + col%16, elem i = row%16.
__device__ __forceinline__ size_t tiled_addr(int c, int col, int nct) {
  return ((size_t)((c >> 5) * nct + (col >> 4)) * 32 +
          (((c >> 4) & 1) * 16 + (col & 15))) * 16 + (c & 15);
}

// ---------------------------------------------------------------------------
// 1) prep: [N,L,64] f32 (real & imag) -> A0 bf16 tiled [128, NCOL]
//    c = p*64 + (k*4+u), col = n*64 + l
// ---------------------------------------------------------------------------
__global__ void prep_kernel(const float* __restrict__ Xr,
                            const float* __restrict__ Xi,
                            bf16_t* __restrict__ A0) {
  __shared__ float tile[64][65];
  const int n = blockIdx.x;
  const int p = blockIdx.y;
  const float* X = p ? Xi : Xr;
  for (int i = threadIdx.x; i < 4096; i += 256) {
    int l = i >> 6, j = i & 63;
    tile[l][j] = X[((size_t)n * 64 + l) * 64 + j];
  }
  __syncthreads();
  for (int i = threadIdx.x; i < 4096; i += 256) {
    int j = i >> 6, l = i & 63;
    A0[tiled_addr(p * 64 + j, n * 64 + l, NCTM)] = (bf16_t)tile[l][j];
  }
}

// ---------------------------------------------------------------------------
// 2) pack weights f32 [M, K] -> bf16 in WMMA A-fragment order:
//    dst[((mt*KT + kt)*32 + lane)*16 + i] = W[m, k]
//    m = mt*16 + (lane&15); k = kt*32 + 16*(lane>>4) + i
// ---------------------------------------------------------------------------
__global__ void packw_kernel(const float* __restrict__ W,
                             bf16_t* __restrict__ dst, int M, int Kdim) {
  int idx = blockIdx.x * 256 + threadIdx.x;
  if (idx >= M * Kdim) return;
  int i     = idx & 15;
  int lane  = (idx >> 4) & 31;
  int tilei = idx >> 9;
  int KT = Kdim >> 5;
  int kt = tilei % KT, mt = tilei / KT;
  int hi = lane >> 4;
  int m  = mt * 16 + (lane & 15);
  int k  = kt * 32 + (hi << 4) + i;
  dst[idx] = (bf16_t)W[(size_t)m * Kdim + k];
}

// ---------------------------------------------------------------------------
// 3) per-(c,n) mean over 64 l columns (tiled in, tiled out). wave32 / group.
// ---------------------------------------------------------------------------
__global__ void colmean_kernel(const bf16_t* __restrict__ Act,
                               bf16_t* __restrict__ Mean, int C) {
  int wid  = (blockIdx.x * blockDim.x + threadIdx.x) >> 5;
  int lane = threadIdx.x & 31;
  if (wid >= C * Nsmp) return;
  int c = wid >> 11, n = wid & (Nsmp - 1);
  int l = lane * 2;
  float s = (float)Act[tiled_addr(c, n * 64 + l, NCTM)] +
            (float)Act[tiled_addr(c, n * 64 + l + 1, NCTM)];
  #pragma unroll
  for (int o = 16; o > 0; o >>= 1) s += __shfl_xor(s, o, 32);
  if (lane == 0)
    Mean[tiled_addr(c, n, Nsmp / 16)] = (bf16_t)(s * (1.0f / 64.0f));
}

// ---------------------------------------------------------------------------
// 4) WMMA GEMM. Wave owns an MT*16-row m-stripe x 16 columns.
//    grid = (ncols/128, M/(MT*16)); block = 256 = 8 waves.
//    A (weights) pre-packed fragment-major (immediate offsets in the loop);
//    B (acts) tiled fragment-major: one running pointer, 32B/lane loads.
//    MT=8 keeps accumulators (64 VGPRs) register-resident (no spills).
// ---------------------------------------------------------------------------
template <int MT>
__global__ void __launch_bounds__(256, 1)
gemm_wmma_kernel(const bf16_t* __restrict__ Wp,
                 const bf16_t* __restrict__ Bm,
                 const float*  __restrict__ Y2, float y2s,
                 int relu,
                 bf16_t* __restrict__ outb,
                 float*  __restrict__ outf,
                 int Kdim, int ncols) {
  const int lane    = threadIdx.x & 31;
  const int wave    = threadIdx.x >> 5;
  const int ct      = blockIdx.x * 8 + wave;   // column tile
  const int mstripe = blockIdx.y;              // m-stripe of MT tiles
  const int nct     = ncols >> 4;
  const int KT      = Kdim >> 5;
  const int hi      = lane >> 4;
  const int nn      = lane & 15;

  v8f acc[MT];
  #pragma unroll
  for (int mt = 0; mt < MT; ++mt) acc[mt] = (v8f){};

  const bf16_t* bptr = Bm + ((size_t)ct * 32 + lane) * 16;
  const size_t  bstride = (size_t)nct * 512;   // elements per k-step
  const bf16_t* aptr = Wp + ((size_t)(mstripe * MT) * KT * 32 + lane) * 16;

  for (int kt = 0; kt < KT; ++kt) {
    v16bf b = *(const v16bf*)bptr;
    if (kt + 1 < KT) __builtin_prefetch(bptr + bstride, 0, 1);
    bptr += bstride;
    #pragma unroll
    for (int mt = 0; mt < MT; ++mt) {
      // constant byte offsets (mt*KT*1024 <= 57KB) -> folded into ioffset
      v16bf a = *(const v16bf*)(aptr + (size_t)mt * KT * 512);
      acc[mt] = __builtin_amdgcn_wmma_f32_16x16x32_bf16(
          false, a, false, b, (short)0, acc[mt], false, false);
    }
    aptr += 512;
  }

  // epilogue: D row m = mstripe*MT*16 + mt*16 + 8*hi + v, column ct*16 + nn
  const int col  = ct * 16 + nn;
  const int nidx = col >> 6;                   // sample index for Y2 broadcast
  #pragma unroll
  for (int mt = 0; mt < MT; ++mt) {
    FragF8 d; d.v = acc[mt];
    #pragma unroll
    for (int v = 0; v < 8; ++v) {
      int m   = mstripe * MT * 16 + mt * 16 + hi * 8 + v;
      float z = d.e[v];
      if (Y2)  z += y2s * Y2[(size_t)m * Nsmp + nidx];
      if (relu) z = fmaxf(z, 0.0f);
      if (outb) outb[tiled_addr(m, col, nct)] = (bf16_t)z;
      else      outf[(size_t)m * ncols + col] = z;
    }
  }
}

// ---------------------------------------------------------------------------
// 5) BatchNorm statistics over tiled activations.
//    Thread t = fixed position within 1KB tiles -> fixed channel.
// ---------------------------------------------------------------------------
#define SCHUNKS 32
__global__ void zero_kernel(float* p, int n) {
  int i = blockIdx.x * 256 + threadIdx.x;
  if (i < n) p[i] = 0.0f;
}

__global__ void stats_kernel(const bf16_t* __restrict__ Act,
                             float* __restrict__ stats, int C) {
  const int kt = blockIdx.x / SCHUNKS;
  const int ch = blockIdx.x % SCHUNKS;
  const int tilesPer = NCTM / SCHUNKS;      // 256
  const int t = threadIdx.x;                // 0..511: position within tile
  const bf16_t* base =
      Act + ((size_t)kt * NCTM + (size_t)ch * tilesPer) * 512 + t;
  float s = 0.f, s2 = 0.f;
  for (int i = 0; i < tilesPer; ++i) {
    float v = (float)base[(size_t)i * 512];
    s += v; s2 += v * v;
  }
  __shared__ float r1[512], r2[512];
  r1[t] = s; r2[t] = s2;
  __syncthreads();
  if (((t >> 4) & 15) == 0) {               // 32 collectors, one per channel
    for (int j = 1; j < 16; ++j) { s += r1[t + 16 * j]; s2 += r2[t + 16 * j]; }
    int c = kt * 32 + ((t >> 8) << 4) + (t & 15);
    atomicAdd(&stats[c], s);
    atomicAdd(&stats[C + c], s2);
  }
}

// ---------------------------------------------------------------------------
// 6) BN apply over tiled buffer (C=256, NCT=8192): channel from flat index.
// ---------------------------------------------------------------------------
__global__ void bn_kernel(const bf16_t* __restrict__ in,
                          bf16_t* __restrict__ out,
                          const float* __restrict__ stats,
                          const float* __restrict__ w,
                          const float* __restrict__ b, int C) {
  size_t idx = (size_t)blockIdx.x * 256 + threadIdx.x;
  int c = (int)(((idx >> 22) << 5) + (((idx >> 8) & 1) << 4) + (idx & 15));
  const float inv = 1.0f / (float)NCOL;
  float mu  = stats[c] * inv;
  float var = stats[C + c] * inv - mu * mu;
  float g   = w[c] * rsqrtf(var + BN_EPS);
  out[idx] = (bf16_t)(((float)in[idx] - mu) * g + b[c]);
}

// ---------------------------------------------------------------------------
// 7) per-(n,k) L2 norm over l; scale = sqrt(L)/||.|| = 8/||.|| (A3 plain f32)
// ---------------------------------------------------------------------------
__global__ void norm_kernel(const float* __restrict__ A3,
                            float* __restrict__ Nrm) {
  int wid  = (blockIdx.x * blockDim.x + threadIdx.x) >> 5;
  int lane = threadIdx.x & 31;
  if (wid >= Nsmp * Ksz) return;
  int k = wid & 15, n = wid >> 4;
  const float* src = A3 + (size_t)k * NCOL + (size_t)n * 64 + lane * 2;
  float s = src[0] * src[0] + src[1] * src[1];
  #pragma unroll
  for (int o = 16; o > 0; o >>= 1) s += __shfl_xor(s, o, 32);
  if (lane == 0) Nrm[n * Ksz + k] = 8.0f * rsqrtf(s);
}

// ---------------------------------------------------------------------------
// 8) Fhat[b,r,l,k,u] = Vhat * A3[k, n*64+l] * Nrm[n,k]
// ---------------------------------------------------------------------------
__global__ void output_kernel(const float* __restrict__ Vhat,
                              const float* __restrict__ A3,
                              const float* __restrict__ Nrm,
                              float* __restrict__ out) {
  size_t i = (size_t)blockIdx.x * 256 + threadIdx.x;   // B*R*L*K*U
  int k = (int)((i >> 2) & 15);
  size_t col = i >> 6;                                  // n*64 + l
  int n = (int)(col >> 6);
  out[i] = Vhat[i] * A3[(size_t)k * NCOL + col] * Nrm[n * Ksz + k];
}

// ---------------------------------------------------------------------------
extern "C" void kernel_launch(void* const* d_in, const int* in_sizes, int n_in,
                              void* d_out, int out_size, void* d_ws, size_t ws_size,
                              hipStream_t stream) {
  (void)in_sizes; (void)n_in; (void)out_size; (void)ws_size;
  const float* Hr   = (const float*)d_in[0];
  const float* Hi   = (const float*)d_in[1];
  const float* Vh   = (const float*)d_in[2];
  const float* P1_0 = (const float*)d_in[3];
  const float* P2_0 = (const float*)d_in[4];
  const float* bnw0 = (const float*)d_in[5];
  const float* bnb0 = (const float*)d_in[6];
  const float* P1_1 = (const float*)d_in[7];
  const float* P2_1 = (const float*)d_in[8];
  const float* bnw1 = (const float*)d_in[9];
  const float* bnb1 = (const float*)d_in[10];
  const float* P1_2 = (const float*)d_in[11];
  const float* P2_2 = (const float*)d_in[12];
  float* out = (float*)d_out;

  // ---- workspace carve-up (256B aligned) ----
  char* ws = (char*)d_ws;
  size_t off = 0;
  auto carve = [&](size_t bytes) {
    size_t o = off;
    off = (off + bytes + 255) & ~(size_t)255;
    return o;
  };
  bf16_t* wA0   = (bf16_t*)(ws + carve((size_t)C0 * NCOL * 2));
  bf16_t* wActA = (bf16_t*)(ws + carve((size_t)C1 * NCOL * 2));
  bf16_t* wActB = (bf16_t*)(ws + carve((size_t)C1 * NCOL * 2));
  float*  wA3   = (float*) (ws + carve((size_t)C3 * NCOL * 4));
  bf16_t* wMean = (bf16_t*)(ws + carve((size_t)C1 * Nsmp * 2));
  float*  wY2   = (float*) (ws + carve((size_t)C1 * Nsmp * 4));
  float*  wSt   = (float*) (ws + carve(2 * C1 * 4));
  float*  wNrm  = (float*) (ws + carve((size_t)Nsmp * Ksz * 4));
  bf16_t* pw10  = (bf16_t*)(ws + carve((size_t)C1 * C0 * 2));
  bf16_t* pw20  = (bf16_t*)(ws + carve((size_t)C1 * C0 * 2));
  bf16_t* pw11  = (bf16_t*)(ws + carve((size_t)C2 * C1 * 2));
  bf16_t* pw21  = (bf16_t*)(ws + carve((size_t)C2 * C1 * 2));
  bf16_t* pw12  = (bf16_t*)(ws + carve((size_t)C3 * C2 * 2));
  bf16_t* pw22  = (bf16_t*)(ws + carve((size_t)C3 * C2 * 2));

  // ---- input prep + weight packing ----
  prep_kernel<<<dim3(Nsmp, 2), 256, 0, stream>>>(Hr, Hi, wA0);
  packw_kernel<<<(C1 * C0 + 255) / 256, 256, 0, stream>>>(P1_0, pw10, C1, C0);
  packw_kernel<<<(C1 * C0 + 255) / 256, 256, 0, stream>>>(P2_0, pw20, C1, C0);
  packw_kernel<<<(C2 * C1 + 255) / 256, 256, 0, stream>>>(P1_1, pw11, C2, C1);
  packw_kernel<<<(C2 * C1 + 255) / 256, 256, 0, stream>>>(P2_1, pw21, C2, C1);
  packw_kernel<<<(C3 * C2 + 255) / 256, 256, 0, stream>>>(P1_2, pw12, C3, C2);
  packw_kernel<<<(C3 * C2 + 255) / 256, 256, 0, stream>>>(P2_2, pw22, C3, C2);

  const dim3 gmain((int)(NCOL / 128), C1 / 128);  // 1024 x 2 (MT=8)
  const dim3 gmean(Nsmp / 128, C1 / 128);         //   16 x 2 (MT=8)
  const dim3 glast((int)(NCOL / 128), 1);         // 1024 x 1 (MT=1)
  const dim3 gml(Nsmp / 128, 1);                  //   16 x 1 (MT=1)

  // ---- layer 0 ----
  colmean_kernel<<<C0 * Nsmp / 8, 256, 0, stream>>>(wA0, wMean, C0);
  gemm_wmma_kernel<8><<<gmean, 256, 0, stream>>>(pw20, wMean, nullptr, 0.f, 0,
                                                 nullptr, wY2, C0, Nsmp);
  gemm_wmma_kernel<8><<<gmain, 256, 0, stream>>>(pw10, wA0, wY2, 0.1f, 1,
                                                 wActA, nullptr, C0, (int)NCOL);
  zero_kernel<<<2, 256, 0, stream>>>(wSt, 2 * C1);
  stats_kernel<<<(C1 / 32) * SCHUNKS, 512, 0, stream>>>(wActA, wSt, C1);
  bn_kernel<<<(int)((size_t)C1 * NCOL / 256), 256, 0, stream>>>(
      wActA, wActB, wSt, bnw0, bnb0, C1);

  // ---- layer 1 ----
  colmean_kernel<<<C1 * Nsmp / 8, 256, 0, stream>>>(wActB, wMean, C1);
  gemm_wmma_kernel<8><<<gmean, 256, 0, stream>>>(pw21, wMean, nullptr, 0.f, 0,
                                                 nullptr, wY2, C1, Nsmp);
  gemm_wmma_kernel<8><<<gmain, 256, 0, stream>>>(pw11, wActB, wY2, 0.1f, 1,
                                                 wActA, nullptr, C1, (int)NCOL);
  zero_kernel<<<2, 256, 0, stream>>>(wSt, 2 * C1);
  stats_kernel<<<(C1 / 32) * SCHUNKS, 512, 0, stream>>>(wActA, wSt, C1);
  bn_kernel<<<(int)((size_t)C1 * NCOL / 256), 256, 0, stream>>>(
      wActA, wActB, wSt, bnw1, bnb1, C1);

  // ---- layer 2 (no relu / no BN, f32 out) ----
  colmean_kernel<<<C1 * Nsmp / 8, 256, 0, stream>>>(wActB, wMean, C1);
  gemm_wmma_kernel<1><<<gml, 256, 0, stream>>>(pw22, wMean, nullptr, 0.f, 0,
                                               nullptr, wY2, C2, Nsmp);
  gemm_wmma_kernel<1><<<glast, 256, 0, stream>>>(pw12, wActB, wY2, 0.1f, 0,
                                                 nullptr, wA3, C2, (int)NCOL);

  // ---- normalization + output ----
  norm_kernel<<<Nsmp * Ksz / 8, 256, 0, stream>>>(wA3, wNrm);
  output_kernel<<<(int)((size_t)Bsz * Rsz * Lsz * Ksz * Usz / 256), 256, 0,
                  stream>>>(Vh, wA3, wNrm, out);
}
